// TemporalAttention_566935683473
// MI455X (gfx1250) — matrix operand order repaired and checked by reading, not compiled, not run on hardware
//
#include <hip/hip_runtime.h>
#include <math.h>

typedef __attribute__((ext_vector_type(16))) __bf16 v16bf;
typedef __attribute__((ext_vector_type(8)))  float  v8f;
typedef __attribute__((ext_vector_type(4)))  float  v4f;

#define T_DIM 512
#define N_DIM 1024
#define D_DIM 256
#define M_TOT (T_DIM * N_DIM)

// Branch-free tanh: v_tanh_f32 if the builtin exists, else exp/rcp formula.
static __device__ __forceinline__ float fast_tanh(float x) {
#if __has_builtin(__builtin_amdgcn_tanhf)
  return __builtin_amdgcn_tanhf(x);
#else
  float ex = __expf(2.0f * x);                       // v_exp_f32 path
  return 1.0f - 2.0f * __builtin_amdgcn_rcpf(ex + 1.0f);
#endif
}

// ---------------------------------------------------------------------------
// Kernel 1: e[t,n] = sum_c tanh( sum_k H[t,n,k] * W[c,k] + b[c] ) * sw[c]
// Block = 512 threads (16 waves); each wave owns a 16-row A tile (bf16 regs).
// W staged in LDS pre-swizzled into B-fragment layout, 64KB per phase.
// ---------------------------------------------------------------------------
__global__ __launch_bounds__(512)
void ta_score_kernel(const float* __restrict__ H, const float* __restrict__ W,
                     const float* __restrict__ b, const float* __restrict__ sw,
                     float* __restrict__ e_out) {
  // [ntl(8)][kb(8)][lane(32)] x 16 bf16 elems = 64 KB
  __shared__ __bf16 Bs[8 * 8 * 32 * 16];

  const int tid  = threadIdx.x;
  const int lane = tid & 31;
  const int wave = tid >> 5;
  const int base_row = blockIdx.x * 256 + wave * 16;
  const int lrow = base_row + (lane & 15);
  const int off0 = (lane & 16) ? 8 : 0;   // A: hi half-wave holds K+8

  // ---- Load 16x256 f32 A tile, convert to 8 bf16 WMMA fragments ----
  v16bf a[8];
  const float* hrow = H + (size_t)lrow * D_DIM;
#pragma unroll
  for (int kb = 0; kb < 8; ++kb) {
    const float* p0 = hrow + kb * 32 + off0;          // 32B aligned
    v4f x0 = *(const v4f*)(p0);                       // K off0+0..3
    v4f x1 = *(const v4f*)(p0 + 4);                   // K off0+4..7
    v4f x2 = *(const v4f*)(p0 + 16);                  // K off0+16..19
    v4f x3 = *(const v4f*)(p0 + 20);                  // K off0+20..23
    v16bf av;
#pragma unroll
    for (int i = 0; i < 4; ++i) {
      av[i]      = (__bf16)x0[i];
      av[4 + i]  = (__bf16)x1[i];
      av[8 + i]  = (__bf16)x2[i];
      av[12 + i] = (__bf16)x3[i];
    }
    a[kb] = av;
  }

  float erow[8];
#pragma unroll
  for (int r = 0; r < 8; ++r) erow[r] = 0.f;

  for (int ph = 0; ph < 2; ++ph) {
    __syncthreads();  // previous phase's readers done before overwrite

    // ---- Stage B: 2048 fragment-vectors of 16 bf16; 4 per thread.
    // Each vector is K {8h..8h+7, 16+8h..16+8h+7} for one column -> two 32B runs in W
    // and contiguous 32B in LDS: vector loads + wide ds stores.
#pragma unroll
    for (int it = 0; it < 4; ++it) {
      int j    = tid + it * 512;           // fragment-vector index
      int bl   = j & 31;                   // lane slot
      int kb   = (j >> 5) & 7;
      int ntl  = (j >> 8) & 7;
      int col  = (ph * 8 + ntl) * 16 + (bl & 15);
      int k0   = kb * 32 + ((bl & 16) ? 8 : 0);
      const float* wp = W + col * D_DIM + k0;
      v4f w0 = *(const v4f*)(wp);
      v4f w1 = *(const v4f*)(wp + 4);
      v4f w2 = *(const v4f*)(wp + 16);
      v4f w3 = *(const v4f*)(wp + 20);
      v16bf bv;
#pragma unroll
      for (int i = 0; i < 4; ++i) {
        bv[i]      = (__bf16)w0[i];
        bv[4 + i]  = (__bf16)w1[i];
        bv[8 + i]  = (__bf16)w2[i];
        bv[12 + i] = (__bf16)w3[i];
      }
      *(v16bf*)(Bs + (size_t)j * 16) = bv;
    }
    __syncthreads();

#pragma unroll
    for (int ntl = 0; ntl < 8; ++ntl) {
      // Preload all 8 B fragments, then issue WMMAs on two acc chains.
      v16bf bv[8];
#pragma unroll
      for (int kb = 0; kb < 8; ++kb)
        bv[kb] = *(const v16bf*)(Bs + (((ntl * 8 + kb) * 32 + lane) * 16));

      v8f acc0 = {}, acc1 = {};
#pragma unroll
      for (int kb = 0; kb < 8; kb += 2) {
        acc0 = __builtin_amdgcn_wmma_f32_16x16x32_bf16(
            false, a[kb],     false, bv[kb],     (short)0, acc0, false, false);
        acc1 = __builtin_amdgcn_wmma_f32_16x16x32_bf16(
            false, a[kb + 1], false, bv[kb + 1], (short)0, acc1, false, false);
      }

      int col    = (ph * 8 + ntl) * 16 + (lane & 15);
      float bias = b[col];
      float swv  = sw[col];
#pragma unroll
      for (int r = 0; r < 8; ++r)
        erow[r] += fast_tanh(acc0[r] + acc1[r] + bias) * swv;
    }
  }

  // Reduce the 16 columns held across each 16-lane half (wave32 shuffles)
#pragma unroll
  for (int r = 0; r < 8; ++r) {
    float v = erow[r];
    v += __shfl_xor(v, 1);
    v += __shfl_xor(v, 2);
    v += __shfl_xor(v, 4);
    v += __shfl_xor(v, 8);
    erow[r] = v;
  }
  if (lane == 0) {
#pragma unroll
    for (int r = 0; r < 8; ++r) e_out[base_row + r] = erow[r];       // rows 0-7
  } else if (lane == 16) {
#pragma unroll
    for (int r = 0; r < 8; ++r) e_out[base_row + 8 + r] = erow[r];   // rows 8-15
  }
}

// ---------------------------------------------------------------------------
// Kernel 2: causal online-softmax scan over t.
// Block = 256 threads covering 4 n-columns x 64 float4 lanes (b128 traffic).
// e column staged in LDS; H load software-pipelined one step ahead.
// ---------------------------------------------------------------------------
__global__ __launch_bounds__(256)
void ta_scan_kernel(const float* __restrict__ H, const float* __restrict__ e,
                    float* __restrict__ C) {
  __shared__ float es[4][T_DIM];          // 8 KB

  const int tid   = threadIdx.x;
  const int nl    = tid >> 6;             // 0..3
  const int d4    = (tid & 63) * 4;       // 0..252
  const int nbase = blockIdx.x * 4;
  const int n     = nbase + nl;

  // Stage e[:, nbase..nbase+3] into LDS
  for (int i = tid; i < 4 * T_DIM; i += 256) {
    int nl2 = i & 3;
    int t   = i >> 2;
    es[nl2][t] = e[t * N_DIM + nbase + nl2];
  }
  __syncthreads();

  const size_t stride_t = (size_t)N_DIM * D_DIM;
  const float* hp = H + (size_t)n * D_DIM + d4;
  float*       cp = C + (size_t)n * D_DIM + d4;

  float m = -INFINITY, s = 0.f;
  v4f num = {};
  v4f hv = *(const v4f*)(hp);             // t = 0 tile

  for (int t = 0; t < T_DIM; ++t) {
    if (t + 8 < T_DIM)
      __builtin_prefetch(hp + (size_t)(t + 8) * stride_t, 0, 3);
    v4f hnext = hv;
    if (t + 1 < T_DIM)
      hnext = *(const v4f*)(hp + (size_t)(t + 1) * stride_t);

    float et    = es[nl][t];
    float mn    = fmaxf(m, et);
    float alpha = __expf(m - mn);         // rescale old accumulators
    float p     = __expf(et - mn);        // weight of current timestep
    s = s * alpha + p;
    float rs = __builtin_amdgcn_rcpf(s);

    v4f out;
#pragma unroll
    for (int i = 0; i < 4; ++i) {
      num[i] = num[i] * alpha + p * hv[i];
      out[i] = num[i] * rs;
    }
    *(v4f*)(cp + (size_t)t * stride_t) = out;

    m  = mn;
    hv = hnext;
  }
}

// ---------------------------------------------------------------------------
extern "C" void kernel_launch(void* const* d_in, const int* in_sizes, int n_in,
                              void* d_out, int out_size, void* d_ws, size_t ws_size,
                              hipStream_t stream) {
  const float* H  = (const float*)d_in[0];   // (T, N, D) f32
  const float* W  = (const float*)d_in[1];   // (D, D)    f32
  const float* bb = (const float*)d_in[2];   // (D,)      f32
  const float* sw = (const float*)d_in[3];   // (D,)      f32
  float* C    = (float*)d_out;               // (T, N, D) f32
  float* e_ws = (float*)d_ws;                // (T*N,)    f32 scratch (2 MB)

  ta_score_kernel<<<M_TOT / 256, 512, 0, stream>>>(H, W, bb, sw, e_ws);
  ta_scan_kernel<<<N_DIM / 4, 256, 0, stream>>>(H, e_ws, C);
}
